// ConvSpikeEncoder_89060441850414
// MI455X (gfx1250) — matrix operand order repaired and checked by reading, not compiled, not run on hardware
//
#include <hip/hip_runtime.h>
#include <hip/hip_bf16.h>

// ---------------- problem constants ----------------
#define B_      32
#define T_      512
#define C_      512
#define HIDDEN_ 512
#define STEPS_  4
#define OUT_    2048
#define M_      (B_ * T_)     // 16384 GEMM rows
#define N_      OUT_          // 2048  GEMM cols
#define K_      C_            // 512   GEMM depth
#define NSTEP_  (T_ * STEPS_) // 2048 LIF steps
#define REC_    ((size_t)NSTEP_ * B_ * HIDDEN_) // elems per record

typedef __attribute__((ext_vector_type(16))) __bf16        v16bf;
typedef __attribute__((ext_vector_type(8)))  float         v8f;
typedef __attribute__((ext_vector_type(4)))  unsigned int  u32x4;
typedef __attribute__((ext_vector_type(8)))  int           i32x8;
typedef __attribute__((ext_vector_type(4)))  int           i32x4;

// LDS tile geometry: TDM pad gives 80B row stride (64B payload + 16B pad)
// -> per-lane ds_load_b128 bank step = 20 banks mod 64: conflict-free.
#define ROWB_    80
#define A_TILE_B (256 * ROWB_)   // 20480 B per buffer
#define B_TILE_B (32  * ROWB_)   //  2560 B per buffer
#define SMEM_B   (2 * A_TILE_B + 2 * B_TILE_B) // 46080 B

// ---------------- tiny utility kernels ----------------
__global__ void zero_f32(float* p, int n) {
    int i = blockIdx.x * blockDim.x + threadIdx.x;
    if (i < n) p[i] = 0.0f;
}

__global__ void cvt_bf16(const float* __restrict__ in, __hip_bfloat16* __restrict__ out, int n) {
    int i = blockIdx.x * blockDim.x + threadIdx.x;
    int stride = gridDim.x * blockDim.x;
    for (; i < n; i += stride) out[i] = __float2bfloat16(in[i]);
}

// ---------------- TDM descriptor build + issue (2D tile, bf16 elements) ----------------
// D# per CDNA5 ISA. Assumes static LDS block starts at
// LDS offset 0 (it is the only __shared__ object in the kernel).
__device__ __forceinline__ void tdm_load_2d(unsigned lds_off, const void* gptr,
                                            unsigned tensor_d0, unsigned tensor_d1,
                                            unsigned tile_d0, unsigned tile_d1,
                                            unsigned stride0) {
    unsigned long long ga = (unsigned long long)gptr;
    u32x4 g0;
    g0[0] = 1u;                                                // count=1, user D#
    g0[1] = lds_off;                                           // LDS byte address
    g0[2] = (unsigned)ga;                                      // global_addr[31:0]
    g0[3] = (unsigned)((ga >> 32) & 0x01FFFFFFu) | 0x80000000u;// addr[56:32] | type=2
    i32x8 g1;
    // data_size=1 (2B), pad_enable, pad_interval=3 (16 DW = 64B), pad_amount=3 (4 DW = 16B)
    g1[0] = (int)((1u << 16) | (1u << 20) | (3u << 22) | (3u << 25));
    g1[1] = (int)(tensor_d0 << 16);                            // tensor_dim0[15:0]
    g1[2] = (int)((tensor_d0 >> 16) | (tensor_d1 << 16));      // dim0 hi | dim1 lo
    g1[3] = (int)((tensor_d1 >> 16) | (tile_d0 << 16));        // dim1 hi | tile_dim0
    g1[4] = (int)tile_d1;                                      // tile_dim1 (tile_dim2=0)
    g1[5] = (int)stride0;                                      // tensor_dim0_stride lo
    g1[6] = 0;                                                 // stride0 hi | stride1 lo
    g1[7] = 0;
    i32x4 z4 = {0, 0, 0, 0};
    i32x8 z8 = {0, 0, 0, 0, 0, 0, 0, 0};
    // 6-arg form on this toolchain: (g0, g1, g2, g3, g4, cpol)
    __builtin_amdgcn_tensor_load_to_lds(g0, g1, z4, z4, z8, 0);
}

// Launder the LDS base pointer: the TDM builtin has no pointer operand, so
// alias analysis cannot see its LDS writes; without this, loads from smem
// fold to undef and the WMMA chain collapses (observed in round 3 asm).
__device__ __forceinline__ const unsigned char* lds_launder(const unsigned char* p) {
    asm volatile("" : "+v"(p) : : "memory");
    return p;
}

// 16B + 16B LDS fragment gather into one WMMA operand
__device__ __forceinline__ v16bf lds_frag(const unsigned char* p, int off0, int off1) {
    union { v16bf v; u32x4 u[2]; } f;
    f.u[0] = *(const u32x4*)(p + off0);
    f.u[1] = *(const u32x4*)(p + off1);
    return f.v;
}

// ---------------- WMMA GEMM: h[M,N] = x[M,K] * w[N,K]^T ----------------
// Block = 128 threads (4 waves). Block tile = 256 rows x 32 cols; wave wv owns
// rows wv*64..wv*64+63 as 4x2 v_wmma_f32_16x16x32_bf16 accumulators.
// A/B K-chunks (32 deep) are staged global->LDS by the Tensor Data Mover,
// double buffered, retired with s_wait_tensorcnt; fragments then come from
// LDS via ds_load_b128 in the ISA 7.12.2 operand layouts:
//   A 16x32: lanes 0-15 rows, K{0..7,16..23}; lanes 16-31 same rows, K{8..15,24..31}
//   B 32x16: lanes 0-15 cols, K{0..7,16..23}; lanes 16-31 same cols, K{8..15,24..31}
__global__ __launch_bounds__(128) void gemm_wmma_tdm(
        const __hip_bfloat16* __restrict__ xb,   // [M_, K_]
        const __hip_bfloat16* __restrict__ wb,   // [N_, K_]
        float* __restrict__ h,                   // [M_, N_]
        float* __restrict__ bn_sum,              // [N_]
        float* __restrict__ bn_sumsq) {          // [N_]
    __shared__ __align__(16) unsigned char smem[SMEM_B];
    const unsigned offA[2] = { 0u, A_TILE_B };
    const unsigned offB[2] = { 2u * A_TILE_B, 2u * A_TILE_B + B_TILE_B };

    const int lane   = threadIdx.x & 31;
    const int wv     = threadIdx.x >> 5;
    const int l15    = lane & 15;
    const int hihalf = lane >> 4;                 // 0 or 1
    const int mBlock = blockIdx.y * 256;          // block row base
    const int mBase  = mBlock + wv * 64;          // wave row base
    const int nBase  = blockIdx.x * 32;           // block col base (uniform)

    v8f acc[4][2];
    #pragma unroll
    for (int i = 0; i < 4; ++i)
        #pragma unroll
        for (int j = 0; j < 2; ++j)
            #pragma unroll
            for (int e = 0; e < 8; ++e) acc[i][j][e] = 0.0f;

    const int NCHUNK = K_ / 32;                   // 16
    // prologue: stage chunk 0
    if (wv == 0) {
        tdm_load_2d(offA[0], xb + (size_t)mBlock * K_, K_, M_, 32, 256, K_);
        tdm_load_2d(offB[0], wb + (size_t)nBase  * K_, K_, N_, 32, 32,  K_);
    }

    // fragment LDS byte offsets (within a tile row of 80B)
    const int aOff0 = hihalf ? 16 : 0;            // K{8..15} : K{0..7}
    const int aOff1 = aOff0 + 32;                 // K{24..31}: K{16..23}
    const int bOff0 = hihalf * 16;                // K8..15  : K0..7
    const int bOff1 = bOff0 + 32;                 // K24..31 : K16..23

    for (int kk = 0; kk < NCHUNK; ++kk) {
        const int cur = kk & 1;
        if (wv == 0) {
            if (kk + 1 < NCHUNK) {
                const int nxt = cur ^ 1;
                const size_t kelem = (size_t)(kk + 1) * 32;
                tdm_load_2d(offA[nxt], xb + (size_t)mBlock * K_ + kelem, K_, M_, 32, 256, K_);
                tdm_load_2d(offB[nxt], wb + (size_t)nBase  * K_ + kelem, K_, N_, 32, 32,  K_);
                __builtin_amdgcn_s_wait_tensorcnt(2); // chunk kk's 2 TDM ops done
            } else {
                __builtin_amdgcn_s_wait_tensorcnt(0);
            }
        }
        __syncthreads();                           // LDS tile 'cur' visible to all waves

        const unsigned char* smbase = lds_launder(smem);

        v16bf af[4], bf[2];
        #pragma unroll
        for (int i = 0; i < 4; ++i) {
            const unsigned char* ap = smbase + offA[cur] + (unsigned)(wv * 64 + i * 16 + l15) * ROWB_;
            af[i] = lds_frag(ap, aOff0, aOff1);
        }
        #pragma unroll
        for (int j = 0; j < 2; ++j) {
            const unsigned char* bp = smbase + offB[cur] + (unsigned)(j * 16 + l15) * ROWB_;
            bf[j] = lds_frag(bp, bOff0, bOff1);
        }

        #pragma unroll
        for (int i = 0; i < 4; ++i)
            #pragma unroll
            for (int j = 0; j < 2; ++j)
                acc[i][j] = __builtin_amdgcn_wmma_f32_16x16x32_bf16(
                    false, af[i], false, bf[j], (short)0, acc[i][j], false, false);

        __syncthreads();                           // reads done before buffer reuse
    }

    // Epilogue: store C tile (lane holds col l15; VGPR r = row hihalf*8 + r)
    // and feed per-channel sum / sum-of-squares for BatchNorm via f32 atomics.
    #pragma unroll
    for (int j = 0; j < 2; ++j) {
        const int col = nBase + j * 16 + l15;
        float s = 0.0f, s2 = 0.0f;
        #pragma unroll
        for (int i = 0; i < 4; ++i) {
            const int rbase = mBase + i * 16 + hihalf * 8;
            #pragma unroll
            for (int r = 0; r < 8; ++r) {
                float v = acc[i][j][r];
                h[(size_t)(rbase + r) * N_ + col] = v;
                s  += v;
                s2 += v * v;
            }
        }
        atomicAdd(&bn_sum[col],   s);
        atomicAdd(&bn_sumsq[col], s2);
    }
}

// ---------------- BN stats -> fused scale/shift (conv bias cancels in BN) ----------------
__global__ void bn_finalize(const float* __restrict__ bn_sum, const float* __restrict__ bn_sumsq,
                            const float* __restrict__ gamma,  const float* __restrict__ beta,
                            float* __restrict__ scaleA, float* __restrict__ shiftB,
                            float* __restrict__ sumspk_slot) {
    int c = blockIdx.x * blockDim.x + threadIdx.x;
    if (c < N_) {
        const float invM = 1.0f / (float)M_;
        float mean = bn_sum[c] * invM;
        float var  = bn_sumsq[c] * invM - mean * mean; // biased variance, as in torch BN
        float inv  = rsqrtf(var + 1e-5f);
        float a    = gamma[c] * inv;
        scaleA[c]  = a;
        shiftB[c]  = beta[c] - mean * a;
    }
    if (blockIdx.x == 0 && threadIdx.x == 0) *sumspk_slot = 0.0f; // d_out is poisoned
}

// ---------------- LIF scan: one thread per (batch, hidden) neuron ----------------
__global__ __launch_bounds__(256) void lif_scan(
        const float* __restrict__ h,            // [B_, T_*OUT_] == [B_, NSTEP_, HIDDEN_]
        const float* __restrict__ scaleA, const float* __restrict__ shiftB,
        const float* __restrict__ lif_beta_p,
        float* __restrict__ spk_rec, float* __restrict__ mem_rec,
        float* __restrict__ sumspk) {
    const int idx = blockIdx.x * blockDim.x + threadIdx.x;   // 0..16383
    const int b = idx >> 9;
    const int c = idx & (HIDDEN_ - 1);
    const float beta = lif_beta_p[0];

    float a4[STEPS_], s4[STEPS_];
    #pragma unroll
    for (int s = 0; s < STEPS_; ++s) {
        const int o = s * HIDDEN_ + c;
        a4[s] = scaleA[o];
        s4[s] = shiftB[o];
    }

    const float* hp = h + (size_t)b * (T_ * OUT_) + c;
    float mem = 0.0f;
    float cnt = 0.0f;
    for (int u = 0; u < NSTEP_; ++u) {
        float hv = hp[(size_t)u * HIDDEN_];
        if (u + 8 < NSTEP_) __builtin_prefetch(hp + (size_t)(u + 8) * HIDDEN_, 0, 0);
        const int s = u & (STEPS_ - 1);
        float v = fmaf(hv, a4[s], s4[s]);                    // BN applied here
        float rst = mem > 1.0f ? 1.0f : 0.0f;                // reset from previous mem
        mem = fmaf(beta, mem, v) - rst;                      // THRESH == 1
        float sp = (mem - 1.0f) > 0.0f ? 1.0f : 0.0f;
        const size_t ob = (size_t)u * (B_ * HIDDEN_) + (size_t)b * HIDDEN_ + c;
        spk_rec[ob] = sp;
        mem_rec[ob] = mem;
        cnt += sp;
    }

    __shared__ float red[256];
    red[threadIdx.x] = cnt;
    __syncthreads();
    #pragma unroll
    for (int off = 128; off > 0; off >>= 1) {
        if ((int)threadIdx.x < off) red[threadIdx.x] += red[threadIdx.x + off];
        __syncthreads();
    }
    if (threadIdx.x == 0) atomicAdd(sumspk, red[0]);
}

// ---------------- launcher ----------------
extern "C" void kernel_launch(void* const* d_in, const int* in_sizes, int n_in,
                              void* d_out, int out_size, void* d_ws, size_t ws_size,
                              hipStream_t stream) {
    const float* x        = (const float*)d_in[0]; // [B,T,C]
    const float* conv_w   = (const float*)d_in[1]; // [OUT,C]
    // d_in[2] = conv_b : per-channel bias cancels exactly in BatchNorm -> unused
    const float* gamma    = (const float*)d_in[3];
    const float* bn_beta  = (const float*)d_in[4];
    const float* lif_beta = (const float*)d_in[5];

    // workspace layout (bytes): xb16 | wb16 | h f32 | sums | fused BN params
    char* ws = (char*)d_ws;
    __hip_bfloat16* xb = (__hip_bfloat16*)ws;                             // 16 MB
    __hip_bfloat16* wb = (__hip_bfloat16*)(ws + (size_t)16777216);        //  2 MB
    float* h        = (float*)(ws + (size_t)18874368);                    // 128 MB
    float* bn_sum   = (float*)(ws + (size_t)153092096);
    float* bn_sumsq = bn_sum + N_;
    float* scaleA   = bn_sumsq + N_;
    float* shiftB   = scaleA + N_;

    float* out     = (float*)d_out;
    float* spk_rec = out;                 // [NSTEP_, B_, HIDDEN_]
    float* mem_rec = out + REC_;          // [NSTEP_, B_, HIDDEN_]
    float* sumspk  = out + (size_t)out_size - 1;

    zero_f32<<<(2 * N_ + 255) / 256, 256, 0, stream>>>(bn_sum, 2 * N_);
    cvt_bf16<<<1024, 256, 0, stream>>>(x, xb, M_ * K_);
    cvt_bf16<<<256, 256, 0, stream>>>(conv_w, wb, N_ * K_);

    gemm_wmma_tdm<<<dim3(N_ / 32, M_ / 256), 128, 0, stream>>>(xb, wb, h, bn_sum, bn_sumsq);

    bn_finalize<<<(N_ + 255) / 256, 256, 0, stream>>>(bn_sum, bn_sumsq, gamma, bn_beta,
                                                      scaleA, shiftB, sumspk);

    lif_scan<<<(B_ * HIDDEN_) / 256, 256, 0, stream>>>(h, scaleA, shiftB, lif_beta,
                                                       spk_rec, mem_rec, sumspk);
}
